// SchNet_3736621548079
// MI455X (gfx1250) — compile-verified, matrix operand order checked by reading, not run on hardware
//
#include <hip/hip_runtime.h>
#include <hip/hip_bf16.h>

// ---------------- problem constants (fixed by the reference) ----------------
#define NA      10000
#define NEDGE   320000
#define HD      128
#define NFILT   128
#define NGAUSS  50
#define NGP     64          // NG padded to a multiple of 32 for WMMA K
#define NLAYER  6
#define NS      32

#define CUTOFF_C 5.0f
#define PI_C     3.14159265358979f
#define GAMMA_C  48.02f                    // 0.5/(CUTOFF/(NG-1))^2
#define OFFSTEP  (CUTOFF_C / (float)(NGAUSS - 1))
#define LOG2_C   0.6931471805599453f

typedef __attribute__((ext_vector_type(16))) _Float16 v16h;
typedef __attribute__((ext_vector_type(8)))  float    v8f;

// ---------------- scalar helpers ----------------
__device__ __forceinline__ float sspf(float x) {
  float sp = (x > 15.f) ? x : __logf(1.f + __expf(x));
  return sp - LOG2_C;
}
__device__ __forceinline__ float sigmf(float x) { return 1.f / (1.f + __expf(-x)); }

// ---------------- WMMA fragment layout (wave32, 16x16x32 f16->f32) ----------
// Fragment K index for half-pair v (0..7), lane-half hi:
//   k32 = ((v&4)<<2) + (hi<<3) + ((v&3)<<1) (+j)
// All fragments are stored "fragment-resident": one 32x16 (B) or 16x32 (A)
// tile = 512 contiguous halves, per-lane chunk of 16 halves at lane*16.
__device__ __forceinline__ int kfrag(int v, int hi) {
  return ((v & 4) << 2) + (hi << 3) + ((v & 3) << 1);
}
// Inverse map: element index within a lane chunk for logical (m, k).
// Returns offset in halves inside a K-major fragment array (512 per kstep).
__device__ __forceinline__ int afrag_idx(int m, int k) {
  int ks = k >> 5, k32 = k & 31;
  int hi = (k32 >> 3) & 1;
  int e = ((k32 & 16) >> 1) | (k32 & 7);   // = 2*v + j
  return ks * 512 + ((m + (hi << 4)) << 4) + e;
}
__device__ __forceinline__ v8f wmma32(v16h a, v16h b, v8f c) {
  return __builtin_amdgcn_wmma_f32_16x16x32_f16(false, a, false, b, (short)0, c,
                                                false, false);
}
// Load one fragment (A or B): per-lane contiguous 32 bytes -> 2x b128 loads.
__device__ __forceinline__ v16h load_frag(const _Float16* t) {
  int lane = threadIdx.x & 31;
  return *(const v16h*)(t + lane * 16);
}
// Gaussian-smearing A fragment computed on the fly: row m = lane&15 has edge
// distance ewm; columns are Gaussian centers (zero-padded past NGAUSS).
__device__ __forceinline__ v16h ea_frag(float ewm, int kb) {
  int lane = threadIdx.x & 31, hi = lane >> 4;
  v16h a;
#pragma unroll
  for (int v = 0; v < 8; ++v) {
    int kk = kb + kfrag(v, hi);
#pragma unroll
    for (int j = 0; j < 2; ++j) {
      int k = kk + j;
      float val = 0.f;
      if (k < NGAUSS) {
        float d = ewm - OFFSTEP * (float)k;
        val = __expf(-GAMMA_C * d * d);
      }
      a[2 * v + j] = (_Float16)val;
    }
  }
  return a;
}

// ------------- weight prep: f32 -> f16 fragment-resident layout -------------
// dst logical shape [K][Nc] (zero-padded), stored as (K/32)*(Nc/16) tiles of
// 512 halves: tile (ks,nb) at ((ks*(Nc/16))+nb)*512, lane-major inside.
__global__ void wprep_frag_k(const float* __restrict__ src, int srows,
                             int scols, _Float16* __restrict__ dst, int K,
                             int Nc, int transp) {
  int i = blockIdx.x * blockDim.x + threadIdx.x;
  if (i >= K * Nc) return;
  int frag = i >> 9;
  int within = i & 511;
  int lane = within >> 4;
  int e = within & 15;
  int nblocks = Nc >> 4;
  int ks = frag / nblocks, nb = frag % nblocks;
  int hi = lane >> 4, n = lane & 15;
  int v = e >> 1, j = e & 1;
  int kk = ks * 32 + kfrag(v, hi) + j;
  int cc = nb * 16 + n;
  float val = 0.f;
  if (!transp) {
    if (kk < srows && cc < scols) val = src[(size_t)kk * scols + cc];
  } else {
    if (cc < srows && kk < scols) val = src[(size_t)cc * scols + kk];
  }
  dst[i] = (_Float16)val;
}

// ---------------- generic node GEMM: C[M,Nc] = op(A[M,K]) @ B_f16[K,Nc] + b --
// block = 32*(Nc/16) threads, one 16-row strip per block, one n-block per wave.
__global__ void __launch_bounds__(256)
node_gemm(const float* __restrict__ A, const _Float16* __restrict__ B,
          const float* __restrict__ bias, float* __restrict__ C,
          int K, int Nc, int sspA) {
  __shared__ __align__(32) _Float16 sA[16 * 128];  // A strip, fragment layout
  int m0 = blockIdx.x * 16;
  for (int t = threadIdx.x; t < 16 * K; t += blockDim.x) {
    int m = t / K, k = t % K;
    float v = A[(size_t)(m0 + m) * K + k];
    if (sspA) v = sspf(v);
    sA[afrag_idx(m, k)] = (_Float16)v;
  }
  __syncthreads();
  int wv = threadIdx.x >> 5;
  int n0 = wv * 16;
  int nblocks = Nc >> 4;
  v8f acc = {};
  for (int ksi = 0; ksi < (K >> 5); ++ksi) {
    v16h a = load_frag(sA + ksi * 512);
    v16h b = load_frag(B + (size_t)(ksi * nblocks + wv) * 512);
    acc = wmma32(a, b, acc);
  }
  int lane = threadIdx.x & 31, n = lane & 15, hi = lane >> 4;
#pragma unroll
  for (int r = 0; r < 8; ++r) {
    float v = acc[r];
    if (bias) v += bias[n0 + n];
    C[(size_t)(m0 + r + hi * 8) * Nc + n0 + n] = v;
  }
}

// ---------------- fused edge forward (one launch per layer) ----------------
// per wave: 16 edges. ea(16x64) @ w1 -> ssp -> @ w2 -> cutoff -> gather p[col]
// -> atomic scatter into agg[row]. No [E,128] arrays ever hit HBM.
__global__ void __launch_bounds__(128)
edge_fwd(const float* __restrict__ ew, const float* __restrict__ p,
         const int* __restrict__ row, const int* __restrict__ col,
         const _Float16* __restrict__ w1p, const float* __restrict__ b1,
         const _Float16* __restrict__ w2, const float* __restrict__ b2,
         float* __restrict__ agg) {
  __shared__ __align__(32) _Float16 stage[4][16 * HD];
  int wv = threadIdx.x >> 5;
  int lane = threadIdx.x & 31;
  int n = lane & 15, hi = lane >> 4;
  int e0 = (blockIdx.x * 4 + wv) * 16;
  float ewm = ew[e0 + n];  // this lane's A-row distance (m = lane&15)

  v16h a0 = ea_frag(ewm, 0);
  v16h a1 = ea_frag(ewm, 32);

  v8f t1[8];
#pragma unroll
  for (int nb = 0; nb < 8; ++nb) {
    v8f acc = {};
    acc = wmma32(a0, load_frag(w1p + (0 * 8 + nb) * 512), acc);
    acc = wmma32(a1, load_frag(w1p + (1 * 8 + nb) * 512), acc);
    t1[nb] = acc;
  }
  _Float16* st = stage[wv];
#pragma unroll
  for (int nb = 0; nb < 8; ++nb) {
    float bb = b1[nb * 16 + n];
#pragma unroll
    for (int r = 0; r < 8; ++r)
      st[afrag_idx(r + hi * 8, nb * 16 + n)] = (_Float16)sspf(t1[nb][r] + bb);
  }
  v8f fc[8];
#pragma unroll
  for (int nb = 0; nb < 8; ++nb) {
    v8f acc = {};
#pragma unroll
    for (int ks = 0; ks < 4; ++ks)
      acc = wmma32(load_frag(st + ks * 512),
                   load_frag(w2 + (size_t)(ks * 8 + nb) * 512), acc);
    fc[nb] = acc;
  }
  float clv = 0.5f * (__cosf(ewm * (PI_C / CUTOFF_C)) + 1.f);
#pragma unroll
  for (int r = 0; r < 8; ++r) {
    float cm = __shfl(clv, r + hi * 8);
    int e = e0 + r + hi * 8;
    int rm = row[e], cc = col[e];
#pragma unroll
    for (int nb = 0; nb < 8; ++nb) {
      int idx = nb * 16 + n;
      float W = cm * (fc[nb][r] + b2[idx]);
      atomicAdd(&agg[(size_t)rm * HD + idx], p[(size_t)cc * HD + idx] * W);
    }
  }
}

// ---------------- fused edge backward (one launch per layer) ---------------
// Recomputes t1/filt (checkpointing), then gW/gp/gfilt -> ga1 -> gt1 -> gea,
// reduces per-edge gew (Gaussian + cutoff chain) with cross-lane shuffles.
__global__ void __launch_bounds__(128)
edge_bwd(const float* __restrict__ ew, const float* __restrict__ gagg,
         const float* __restrict__ p, const int* __restrict__ row,
         const int* __restrict__ col, const _Float16* __restrict__ w1p,
         const float* __restrict__ b1, const _Float16* __restrict__ w2,
         const float* __restrict__ b2, const _Float16* __restrict__ w2t,
         const _Float16* __restrict__ w1pt, float* __restrict__ gp,
         float* __restrict__ gew) {
  __shared__ __align__(32) _Float16 stage[4][16 * HD];
  int wv = threadIdx.x >> 5;
  int lane = threadIdx.x & 31;
  int n = lane & 15, hi = lane >> 4;
  int e0 = (blockIdx.x * 4 + wv) * 16;
  float ewm = ew[e0 + n];

  v16h a0 = ea_frag(ewm, 0);
  v16h a1 = ea_frag(ewm, 32);

  // t1 = ea @ w1 + b1 (kept for sigmoid), a1 = ssp(t1) staged for GEMM2
  v8f t1[8];
  _Float16* st = stage[wv];
#pragma unroll
  for (int nb = 0; nb < 8; ++nb) {
    v8f acc = {};
    acc = wmma32(a0, load_frag(w1p + (0 * 8 + nb) * 512), acc);
    acc = wmma32(a1, load_frag(w1p + (1 * 8 + nb) * 512), acc);
    float bb = b1[nb * 16 + n];
#pragma unroll
    for (int r = 0; r < 8; ++r) {
      acc[r] += bb;
      st[afrag_idx(r + hi * 8, nb * 16 + n)] = (_Float16)sspf(acc[r]);
    }
    t1[nb] = acc;
  }
  // filt = ssp(t1) @ w2 (+ b2 in epilogue)
  v8f fc[8];
#pragma unroll
  for (int nb = 0; nb < 8; ++nb) {
    v8f acc = {};
#pragma unroll
    for (int ks = 0; ks < 4; ++ks)
      acc = wmma32(load_frag(st + ks * 512),
                   load_frag(w2 + (size_t)(ks * 8 + nb) * 512), acc);
    fc[nb] = acc;
  }
  float clv = 0.5f * (__cosf(ewm * (PI_C / CUTOFF_C)) + 1.f);

  float gcacc[8];
#pragma unroll
  for (int r = 0; r < 8; ++r) gcacc[r] = 0.f;

  // gmsg = gagg[row]; gW = gmsg*p[col]; gp[col] += gmsg*W; gfilt = c*gW -> LDS
#pragma unroll
  for (int r = 0; r < 8; ++r) {
    float cm = __shfl(clv, r + hi * 8);
    int e = e0 + r + hi * 8;
    int rm = row[e], cc = col[e];
#pragma unroll
    for (int nb = 0; nb < 8; ++nb) {
      int idx = nb * 16 + n;
      float filt = fc[nb][r] + b2[idx];
      float gm = gagg[(size_t)rm * HD + idx];
      float gW = gm * p[(size_t)cc * HD + idx];
      atomicAdd(&gp[(size_t)cc * HD + idx], gm * cm * filt);
      gcacc[r] += gW * filt;
      st[afrag_idx(r + hi * 8, idx)] = (_Float16)(cm * gW);  // gfilt
    }
  }
  // ga1 = gfilt @ w2^T
  v8f ga[8];
#pragma unroll
  for (int nb = 0; nb < 8; ++nb) {
    v8f acc = {};
#pragma unroll
    for (int ks = 0; ks < 4; ++ks)
      acc = wmma32(load_frag(st + ks * 512),
                   load_frag(w2t + (size_t)(ks * 8 + nb) * 512), acc);
    ga[nb] = acc;
  }
  // gt1 = ga1 * sigmoid(t1) -> LDS (reuse stage)
#pragma unroll
  for (int nb = 0; nb < 8; ++nb)
#pragma unroll
    for (int r = 0; r < 8; ++r)
      st[afrag_idx(r + hi * 8, nb * 16 + n)] =
          (_Float16)(ga[nb][r] * sigmf(t1[nb][r]));
  // gea = gt1 @ w1^T  (16x64), w1pt has 4 n-blocks
  v8f ge[4];
#pragma unroll
  for (int nb = 0; nb < 4; ++nb) {
    v8f acc = {};
#pragma unroll
    for (int ks = 0; ks < 4; ++ks)
      acc = wmma32(load_frag(st + ks * 512),
                   load_frag(w1pt + (size_t)(ks * 4 + nb) * 512), acc);
    ge[nb] = acc;
  }
  // gew[e] += sum_k gea * d(ea)/d(ew) + gc * d(c)/d(ew)
#pragma unroll
  for (int r = 0; r < 8; ++r) {
    float ewr = __shfl(ewm, r + hi * 8);
    float s = 0.f;
#pragma unroll
    for (int nb = 0; nb < 4; ++nb) {
      int k = nb * 16 + n;
      if (k < NGAUSS) {
        float d = ewr - OFFSTEP * (float)k;
        s += ge[nb][r] * (-2.f * GAMMA_C * d) * __expf(-GAMMA_C * d * d);
      }
    }
    float gcs = gcacc[r];
#pragma unroll
    for (int o = 1; o < 16; o <<= 1) {
      s += __shfl_xor(s, o);
      gcs += __shfl_xor(gcs, o);
    }
    float dcd = -0.5f * (PI_C / CUTOFF_C) * __sinf(ewr * (PI_C / CUTOFF_C));
    float tot = s + gcs * dcd;
    if (n == 0) gew[e0 + r + hi * 8] += tot;  // exclusive owner, no atomic
  }
}

// ---------------- small elementwise / head kernels ----------------
__global__ void embed_k(const int* __restrict__ z, const float* __restrict__ emb,
                        float* __restrict__ h) {
  int i = blockIdx.x * blockDim.x + threadIdx.x;
  if (i < NA * HD) h[i] = emb[(size_t)z[i >> 7] * HD + (i & 127)];
}
__global__ void edge_geom_k(const float* __restrict__ pos,
                            const int* __restrict__ row,
                            const int* __restrict__ col,
                            float* __restrict__ ew) {
  int e = blockIdx.x * blockDim.x + threadIdx.x;
  if (e >= NEDGE) return;
  int r = row[e], c = col[e];
  float dx = pos[r * 3 + 0] - pos[c * 3 + 0];
  float dy = pos[r * 3 + 1] - pos[c * 3 + 1];
  float dz = pos[r * 3 + 2] - pos[c * 3 + 2];
  ew[e] = sqrtf(dx * dx + dy * dy + dz * dz);
}
__global__ void gt3_k(const float* __restrict__ t3, const float* __restrict__ w2,
                      float* __restrict__ g64) {
  int i = blockIdx.x * blockDim.x + threadIdx.x;
  if (i < NA * 64) g64[i] = sigmf(t3[i]) * w2[i & 63];
}
__global__ void gt2_k(float* __restrict__ ga, const float* __restrict__ t2) {
  int i = blockIdx.x * blockDim.x + threadIdx.x;
  if (i < NA * HD) ga[i] *= sigmf(t2[i]);
}
__global__ void energy_k(const float* __restrict__ t3,
                         const float* __restrict__ w2,
                         const float* __restrict__ b2,
                         const int* __restrict__ sidx,
                         float* __restrict__ energy) {
  int a = blockIdx.x * blockDim.x + threadIdx.x;
  if (a >= NA) return;
  float s = b2[0];
  for (int k = 0; k < 64; ++k) s += sspf(t3[(size_t)a * 64 + k]) * w2[k];
  atomicAdd(&energy[sidx[a]], s);
}
__global__ void stress_k(const float* __restrict__ t3,
                         const float* __restrict__ w2,
                         const float* __restrict__ b2,
                         const int* __restrict__ sidx,
                         float* __restrict__ stress) {
  int a = blockIdx.x * blockDim.x + threadIdx.x;
  if (a >= NA) return;
  float act[64];
  for (int k = 0; k < 64; ++k) act[k] = sspf(t3[(size_t)a * 64 + k]);
  int si = sidx[a];
  for (int j = 0; j < 6; ++j) {
    float s = b2[j];
    for (int k = 0; k < 64; ++k) s += act[k] * w2[k * 6 + j];
    atomicAdd(&stress[si * 6 + j], s);
  }
}
__global__ void force_k(const float* __restrict__ pos,
                        const int* __restrict__ row,
                        const int* __restrict__ col,
                        const float* __restrict__ gew, float* __restrict__ F) {
  int e = blockIdx.x * blockDim.x + threadIdx.x;
  if (e >= NEDGE) return;
  int r = row[e], c = col[e];
  float dx = pos[r * 3 + 0] - pos[c * 3 + 0];
  float dy = pos[r * 3 + 1] - pos[c * 3 + 1];
  float dz = pos[r * 3 + 2] - pos[c * 3 + 2];
  float d = sqrtf(dx * dx + dy * dy + dz * dz);
  float g = (d > 1e-12f) ? gew[e] / d : 0.f;
  atomicAdd(&F[r * 3 + 0], -g * dx);
  atomicAdd(&F[r * 3 + 1], -g * dy);
  atomicAdd(&F[r * 3 + 2], -g * dz);
  atomicAdd(&F[c * 3 + 0], g * dx);
  atomicAdd(&F[c * 3 + 1], g * dy);
  atomicAdd(&F[c * 3 + 2], g * dz);
}

// ---------------- host orchestration ----------------
static inline int ceil_div(int a, int b) { return (a + b - 1) / b; }

extern "C" void kernel_launch(void* const* d_in, const int* in_sizes, int n_in,
                              void* d_out, int out_size, void* d_ws,
                              size_t ws_size, hipStream_t stream) {
  (void)in_sizes; (void)n_in; (void)ws_size;
  const int*   z      = (const int*)d_in[0];
  const float* pos    = (const float*)d_in[1];
  const int*   eidx   = (const int*)d_in[2];
  const int*   row    = eidx;
  const int*   col    = eidx + NEDGE;
  const int*   sidx   = (const int*)d_in[3];
  const float* emb    = (const float*)d_in[4];
  const float* mlp_w1 = (const float*)d_in[5];
  const float* mlp_b1 = (const float*)d_in[6];
  const float* mlp_w2 = (const float*)d_in[7];
  const float* mlp_b2 = (const float*)d_in[8];
  const float* lin1_w = (const float*)d_in[9];
  const float* lin2_w = (const float*)d_in[10];
  const float* lin2_b = (const float*)d_in[11];
  const float* lin_w  = (const float*)d_in[12];
  const float* lin_b  = (const float*)d_in[13];
  const float* eh_w1  = (const float*)d_in[14];
  const float* eh_b1  = (const float*)d_in[15];
  const float* eh_w2  = (const float*)d_in[16];
  const float* eh_b2  = (const float*)d_in[17];
  const float* sh_w1  = (const float*)d_in[18];
  const float* sh_b1  = (const float*)d_in[19];
  const float* sh_w2  = (const float*)d_in[20];
  const float* sh_b2  = (const float*)d_in[21];

  float* out    = (float*)d_out;
  float* forces = out;                 // [NA,3]
  float* energy = out + NA * 3;        // [NS]
  float* stress = out + NA * 3 + NS;   // [NS,6]

  // -------- workspace carving (f32 first, then 32B-aligned f16) --------
  const size_t NH = (size_t)NA * HD;
  float* fp = (float*)d_ws;
  auto takeF = [&](size_t nel) { float* r = fp; fp += nel; return r; };
  float* hbuf  = takeF(NH * 7);            // h_0..h_6
  float* t2buf = takeF(NH * 6);            // pre-activation t2 per layer
  float* pbuf  = takeF(NH);
  float* aggb  = takeF(NH);
  float* gbuf  = takeF(NH);
  float* gab   = takeF(NH);
  float* gaggb = takeF(NH);
  float* gpb   = takeF(NH);
  float* t3e   = takeF((size_t)NA * 64);
  float* t3s   = takeF((size_t)NA * 64);
  float* g64   = takeF((size_t)NA * 64);
  float* ewb   = takeF(NEDGE);
  float* gewb  = takeF(NEDGE);
  _Float16* hp = (_Float16*)fp;            // byte offset is 32B-aligned here
  auto takeH = [&](size_t nel) { _Float16* r = hp; hp += nel; return r; };
  const size_t S2 = (size_t)HD * HD;
  _Float16* w1ph   = takeH((size_t)NLAYER * NGP * HD);
  _Float16* w1pth  = takeH((size_t)NLAYER * HD * NGP);
  _Float16* w2h    = takeH((size_t)NLAYER * S2);
  _Float16* w2th   = takeH((size_t)NLAYER * S2);
  _Float16* l1h    = takeH((size_t)NLAYER * S2);
  _Float16* l1th   = takeH((size_t)NLAYER * S2);
  _Float16* l2h    = takeH((size_t)NLAYER * S2);
  _Float16* l2th   = takeH((size_t)NLAYER * S2);
  _Float16* lwh    = takeH((size_t)NLAYER * S2);
  _Float16* lwth   = takeH((size_t)NLAYER * S2);
  _Float16* ehw1h  = takeH((size_t)HD * 64);
  _Float16* ehw1th = takeH((size_t)64 * HD);
  _Float16* shw1h  = takeH((size_t)HD * 64);

  // -------- weight prep (f32 -> f16 fragment-resident, transposes, pad) ----
  auto prep = [&](const float* src, int sr, int sc, _Float16* dst, int K,
                  int Nc, int tr) {
    wprep_frag_k<<<ceil_div(K * Nc, 256), 256, 0, stream>>>(src, sr, sc, dst,
                                                            K, Nc, tr);
  };
  for (int i = 0; i < NLAYER; ++i) {
    prep(mlp_w1 + (size_t)i * NGAUSS * NFILT, NGAUSS, NFILT,
         w1ph + (size_t)i * NGP * HD, NGP, HD, 0);
    prep(mlp_w1 + (size_t)i * NGAUSS * NFILT, NGAUSS, NFILT,
         w1pth + (size_t)i * HD * NGP, HD, NGP, 1);
    prep(mlp_w2 + i * S2, NFILT, NFILT, w2h + i * S2, HD, HD, 0);
    prep(mlp_w2 + i * S2, NFILT, NFILT, w2th + i * S2, HD, HD, 1);
    prep(lin1_w + i * S2, HD, NFILT, l1h + i * S2, HD, HD, 0);
    prep(lin1_w + i * S2, HD, NFILT, l1th + i * S2, HD, HD, 1);
    prep(lin2_w + i * S2, NFILT, HD, l2h + i * S2, HD, HD, 0);
    prep(lin2_w + i * S2, NFILT, HD, l2th + i * S2, HD, HD, 1);
    prep(lin_w + i * S2, HD, HD, lwh + i * S2, HD, HD, 0);
    prep(lin_w + i * S2, HD, HD, lwth + i * S2, HD, HD, 1);
  }
  prep(eh_w1, HD, 64, ehw1h, HD, 64, 0);
  prep(eh_w1, HD, 64, ehw1th, 64, HD, 1);
  prep(sh_w1, HD, 64, shw1h, HD, 64, 0);

  // -------- forward --------
  hipMemsetAsync(d_out, 0, (size_t)out_size * sizeof(float), stream);
  embed_k<<<ceil_div(NA * HD, 256), 256, 0, stream>>>(z, emb, hbuf);
  edge_geom_k<<<ceil_div(NEDGE, 256), 256, 0, stream>>>(pos, row, col, ewb);

  for (int i = 0; i < NLAYER; ++i) {
    float* hcur  = hbuf + (size_t)i * NH;
    float* hnext = hbuf + (size_t)(i + 1) * NH;
    float* t2    = t2buf + (size_t)i * NH;
    node_gemm<<<NA / 16, 256, 0, stream>>>(hcur, l1h + i * S2, nullptr, pbuf,
                                           HD, HD, 0);               // p = h@lin1
    hipMemsetAsync(aggb, 0, NH * sizeof(float), stream);
    edge_fwd<<<NEDGE / 64, 128, 0, stream>>>(
        ewb, pbuf, row, col, w1ph + (size_t)i * NGP * HD, mlp_b1 + i * NFILT,
        w2h + i * S2, mlp_b2 + i * NFILT, aggb);
    node_gemm<<<NA / 16, 256, 0, stream>>>(aggb, l2h + i * S2, lin2_b + i * HD,
                                           t2, HD, HD, 0);           // t2
    node_gemm<<<NA / 16, 256, 0, stream>>>(t2, lwh + i * S2, lin_b + i * HD,
                                           hnext, HD, HD, 1);        // ssp(t2)@lin
  }
  float* hfin = hbuf + (size_t)NLAYER * NH;
  node_gemm<<<NA / 16, 128, 0, stream>>>(hfin, ehw1h, eh_b1, t3e, HD, 64, 0);
  node_gemm<<<NA / 16, 128, 0, stream>>>(hfin, shw1h, sh_b1, t3s, HD, 64, 0);
  energy_k<<<ceil_div(NA, 256), 256, 0, stream>>>(t3e, eh_w2, eh_b2, sidx,
                                                  energy);
  stress_k<<<ceil_div(NA, 128), 128, 0, stream>>>(t3s, sh_w2, sh_b2, sidx,
                                                  stress);

  // -------- backward (d sum(energy) / d positions) --------
  gt3_k<<<ceil_div(NA * 64, 256), 256, 0, stream>>>(t3e, eh_w2, g64);
  node_gemm<<<NA / 16, 256, 0, stream>>>(g64, ehw1th, nullptr, gbuf, 64, HD, 0);
  hipMemsetAsync(gewb, 0, NEDGE * sizeof(float), stream);
  for (int i = NLAYER - 1; i >= 0; --i) {
    float* hcur = hbuf + (size_t)i * NH;
    float* t2   = t2buf + (size_t)i * NH;
    node_gemm<<<NA / 16, 256, 0, stream>>>(gbuf, lwth + i * S2, nullptr, gab,
                                           HD, HD, 0);               // ga2
    gt2_k<<<ceil_div(NA * HD, 256), 256, 0, stream>>>(gab, t2);      // gt2
    node_gemm<<<NA / 16, 256, 0, stream>>>(gab, l2th + i * S2, nullptr, gaggb,
                                           HD, HD, 0);               // gagg
    node_gemm<<<NA / 16, 256, 0, stream>>>(hcur, l1h + i * S2, nullptr, pbuf,
                                           HD, HD, 0);               // p recompute
    hipMemsetAsync(gpb, 0, NH * sizeof(float), stream);
    edge_bwd<<<NEDGE / 64, 128, 0, stream>>>(
        ewb, gaggb, pbuf, row, col, w1ph + (size_t)i * NGP * HD,
        mlp_b1 + i * NFILT, w2h + i * S2, mlp_b2 + i * NFILT, w2th + i * S2,
        w1pth + (size_t)i * HD * NGP, gpb, gewb);
    node_gemm<<<NA / 16, 256, 0, stream>>>(gpb, l1th + i * S2, nullptr, gbuf,
                                           HD, HD, 0);               // g prev
  }
  force_k<<<ceil_div(NEDGE, 256), 256, 0, stream>>>(pos, row, col, gewb,
                                                    forces);
}
